// ExpertLayer_85847806312832
// MI455X (gfx1250) — matrix operand order repaired and checked by reading, not compiled, not run on hardware
//
#include <hip/hip_runtime.h>

// Problem constants (B=4, S=2048, H=512, E=8, F=512)
#define N_TOK 8192
#define HDIM  512
#define NEXP  8
#define FDIM  512

typedef __attribute__((ext_vector_type(8)))  float          v8f;
typedef __attribute__((ext_vector_type(16))) __bf16         v16bf;
typedef __attribute__((ext_vector_type(16))) unsigned short v16u;
typedef __attribute__((ext_vector_type(8)))  unsigned short v8u;

union ABFrag { v16u u; v8u h[2]; v16bf bf; };

__device__ __forceinline__ unsigned short f2bf(float f) {
  union { float f; unsigned u; } v; v.f = f;
  unsigned r = v.u + 0x7FFFu + ((v.u >> 16) & 1u);   // RNE fp32->bf16
  return (unsigned short)(r >> 16);
}

// ---------------------------------------------------------------- init
__global__ void moe_init(int* cnt, int* cursor) {
  int t = threadIdx.x;
  if (t < NEXP) { cnt[t] = 0; cursor[t] = 0; }
}

// ---------------------------------------------------------------- weight pack (fp32 -> bf16 WMMA B-fragment layout)
// Packed index: (((e*32 + nt)*16 + ks)*512 + lane*16 + j)
//   fragment = 32x16 bf16 B tile for K-step ks, N-tile nt of expert e
//   lane l holds column n = l&15; element j -> K = ks*32 + (j>>3)*16 + (l>>4)*8 + (j&7)
__global__ void moe_pack(const float* __restrict__ W1, const float* __restrict__ W2,
                         unsigned short* __restrict__ W1p, unsigned short* __restrict__ W2p) {
  long gid = (long)blockIdx.x * blockDim.x + threadIdx.x;
  const long per = (long)NEXP * 32 * 16 * 512;   // 2M elements per matrix
  const float* src = W1; unsigned short* dst = W1p;
  long p = gid;
  if (gid >= per) { src = W2; dst = W2p; p = gid - per; }
  int j    = (int)(p & 15);
  int lane = (int)((p >> 4) & 31);
  long rest = p >> 9;
  int ks = (int)(rest & 15);
  int nt = (int)((rest >> 4) & 31);
  int e  = (int)(rest >> 9);
  int k = ks * 32 + ((j >> 3) << 4) + ((lane >> 4) << 3) + (j & 7);   // K index (contraction dim)
  int n = nt * 16 + (lane & 15);              // N column
  // W1:(E,H,F) K=H,N=F ; W2:(E,F,H) K=F,N=H  -> same arithmetic since H==F==512
  dst[p] = f2bf(src[((long)e * 512 + k) * 512 + n]);
}

// ---------------------------------------------------------------- router: one wave32 per token
__global__ void moe_router(const float* __restrict__ x, const float* __restrict__ Wr,
                           const float* __restrict__ br, float* __restrict__ gate,
                           int* __restrict__ idxArr, int* __restrict__ cnt) {
  int wave = threadIdx.x >> 5;
  int lane = threadIdx.x & 31;
  int tok = blockIdx.x * 8 + wave;
  float acc[NEXP];
#pragma unroll
  for (int e = 0; e < NEXP; ++e) acc[e] = 0.f;
  const float* xr = x + (size_t)tok * HDIM;
  for (int h = lane; h < HDIM; h += 32) {
    float xv = xr[h];
    const float4* w = (const float4*)(Wr + (size_t)h * NEXP);
    float4 w0 = w[0], w1 = w[1];
    acc[0] += xv * w0.x; acc[1] += xv * w0.y; acc[2] += xv * w0.z; acc[3] += xv * w0.w;
    acc[4] += xv * w1.x; acc[5] += xv * w1.y; acc[6] += xv * w1.z; acc[7] += xv * w1.w;
  }
#pragma unroll
  for (int e = 0; e < NEXP; ++e) {
#pragma unroll
    for (int s = 16; s; s >>= 1) acc[e] += __shfl_xor(acc[e], s, 32);
  }
  if (lane == 0) {
    float lg[NEXP];
#pragma unroll
    for (int e = 0; e < NEXP; ++e) lg[e] = acc[e] + br[e];
    float mx = lg[0]; int mi = 0;
#pragma unroll
    for (int e = 1; e < NEXP; ++e) if (lg[e] > mx) { mx = lg[e]; mi = e; }
    float s = 0.f;
#pragma unroll
    for (int e = 0; e < NEXP; ++e) s += __expf(lg[e] - mx);
    gate[tok] = 1.f / s;       // softmax value at the argmax == top-1 gate
    idxArr[tok] = mi;
    atomicAdd(&cnt[mi], 1);
  }
}

// ---------------------------------------------------------------- scan (E==8, trivial serial)
__global__ void moe_scan(const int* __restrict__ cnt, int* off, int* tileOff) {
  if (threadIdx.x == 0) {
    int o = 0, t = 0;
    off[0] = 0; tileOff[0] = 0;
    for (int e = 0; e < NEXP; ++e) {
      o += cnt[e];             off[e + 1] = o;
      t += (cnt[e] + 63) >> 6; tileOff[e + 1] = t;
    }
  }
}

// ---------------------------------------------------------------- scatter tokens into expert buckets
__global__ void moe_scatter(const int* __restrict__ idxArr, const int* __restrict__ off,
                            int* cursor, int* __restrict__ perm) {
  int t = blockIdx.x * blockDim.x + threadIdx.x;
  if (t < N_TOK) {
    int e = idxArr[t];
    int p = atomicAdd(&cursor[e], 1);
    perm[off[e] + p] = t;
  }
}

// ---------------------------------------------------------------- fused grouped expert GEMM (bf16 WMMA)
#define LDW 520   // bf16 row stride: 1040B = 16B aligned, 1040%256==16 -> conflict-free b128 A loads

__global__ void __launch_bounds__(256)
moe_expert_gemm(const float* __restrict__ x,
                const float* __restrict__ b1, const float* __restrict__ b2,
                const unsigned short* __restrict__ W1p, const unsigned short* __restrict__ W2p,
                const float* __restrict__ gate, const int* __restrict__ perm,
                const int* __restrict__ cnt, const int* __restrict__ off,
                const int* __restrict__ tileOff, float* __restrict__ out) {
  __shared__ unsigned short lds[64 * LDW];   // X tile (bf16), reused for h1 (bf16)
  __shared__ int   tokL[64];
  __shared__ float gateL[64];

  int blk = blockIdx.x;
  if (blk >= tileOff[NEXP]) return;
  int e = 0;
#pragma unroll
  for (int i = 1; i < NEXP; ++i) if (blk >= tileOff[i]) e = i;
  int lb = blk - tileOff[e];
  int rowStart = off[e] + lb * 64;
  int valid = cnt[e] - lb * 64;
  if (valid > 64) valid = 64;

  int tid = threadIdx.x;
  for (int i = tid; i < 64; i += 256) {
    int tk = -1; float g = 0.f;
    if (i < valid) { tk = perm[rowStart + i]; g = gate[tk]; }
    tokL[i] = tk; gateL[i] = g;
  }
  __syncthreads();
  // stage gathered X rows as bf16 (coalesced fp32 global reads)
  for (int i = tid; i < 64 * 512; i += 256) {
    int r = i >> 9, c = i & 511;
    float v = (tokL[r] >= 0) ? x[(size_t)tokL[r] * HDIM + c] : 0.f;
    lds[r * LDW + c] = f2bf(v);
  }
  __syncthreads();

  int wave = tid >> 5, lane = tid & 31;
  int mrow = lane & 15;           // A: row within 16-row tile
  int kb   = (lane >> 4) << 3;    // A: 8-wide K sub-block select
  int nBase = wave * 64;          // this wave's 64 output columns
  int ncol = lane & 15;           // C/D: column within tile
  int cOff = (lane >> 4) << 3;    // C/D: +8 row offset for upper lanes

  const v8f vzero = {0, 0, 0, 0, 0, 0, 0, 0};
  v8f acc[4][4];
#pragma unroll
  for (int a = 0; a < 4; ++a)
#pragma unroll
    for (int b = 0; b < 4; ++b) acc[a][b] = vzero;

  // ---- GEMM1: h1 = relu(X @ W1[e] + b1[e]),  K = H = 512
  // unroll-by-2: one iteration's 16 WMMAs hide the next iteration's B loads
#pragma unroll 2
  for (int ks = 0; ks < 16; ++ks) {
    int k0 = ks * 32;
    ABFrag A[4];
#pragma unroll
    for (int mt = 0; mt < 4; ++mt) {
      int row = mt * 16 + mrow;
      A[mt].h[0] = *(const v8u*)&lds[row * LDW + k0 + kb];
      A[mt].h[1] = *(const v8u*)&lds[row * LDW + k0 + 16 + kb];
    }
#pragma unroll
    for (int ntl = 0; ntl < 4; ++ntl) {
      ABFrag Bf;
      size_t base = ((((size_t)e * 32) + (size_t)(nBase >> 4) + ntl) * 16 + ks) * 512 + lane * 16;
      Bf.u = *(const v16u*)&W1p[base];
#pragma unroll
      for (int mt = 0; mt < 4; ++mt)
        acc[mt][ntl] = __builtin_amdgcn_wmma_f32_16x16x32_bf16(
            false, A[mt].bf, false, Bf.bf, (short)0, acc[mt][ntl], false, false);
    }
  }
  __syncthreads();   // all X reads done -> safe to overwrite LDS with h1
#pragma unroll
  for (int ntl = 0; ntl < 4; ++ntl) {
    int col = nBase + ntl * 16 + ncol;
    float bv = b1[e * FDIM + col];
#pragma unroll
    for (int mt = 0; mt < 4; ++mt)
#pragma unroll
      for (int r = 0; r < 8; ++r) {
        float v = acc[mt][ntl][r] + bv;
        v = v > 0.f ? v : 0.f;
        lds[(mt * 16 + cOff + r) * LDW + col] = f2bf(v);
      }
  }
  __syncthreads();

  // ---- GEMM2: y = (h1 @ W2[e] + b2[e]) * gate,  K = F = 512
#pragma unroll
  for (int a = 0; a < 4; ++a)
#pragma unroll
    for (int b = 0; b < 4; ++b) acc[a][b] = vzero;
#pragma unroll 2
  for (int ks = 0; ks < 16; ++ks) {
    int k0 = ks * 32;
    ABFrag A[4];
#pragma unroll
    for (int mt = 0; mt < 4; ++mt) {
      int row = mt * 16 + mrow;
      A[mt].h[0] = *(const v8u*)&lds[row * LDW + k0 + kb];
      A[mt].h[1] = *(const v8u*)&lds[row * LDW + k0 + 16 + kb];
    }
#pragma unroll
    for (int ntl = 0; ntl < 4; ++ntl) {
      ABFrag Bf;
      size_t base = ((((size_t)e * 32) + (size_t)(nBase >> 4) + ntl) * 16 + ks) * 512 + lane * 16;
      Bf.u = *(const v16u*)&W2p[base];
#pragma unroll
      for (int mt = 0; mt < 4; ++mt)
        acc[mt][ntl] = __builtin_amdgcn_wmma_f32_16x16x32_bf16(
            false, A[mt].bf, false, Bf.bf, (short)0, acc[mt][ntl], false, false);
    }
  }
  // gated scatter-store
#pragma unroll
  for (int ntl = 0; ntl < 4; ++ntl) {
    int col = nBase + ntl * 16 + ncol;
    float bv = b2[e * HDIM + col];
#pragma unroll
    for (int mt = 0; mt < 4; ++mt)
#pragma unroll
      for (int r = 0; r < 8; ++r) {
        int m = mt * 16 + cOff + r;
        int tk = tokL[m];
        if (tk >= 0)
          out[(size_t)tk * HDIM + col] = (acc[mt][ntl][r] + bv) * gateL[m];
      }
  }
}

// ---------------------------------------------------------------- host launcher
extern "C" void kernel_launch(void* const* d_in, const int* in_sizes, int n_in,
                              void* d_out, int out_size, void* d_ws, size_t ws_size,
                              hipStream_t stream) {
  const float* x  = (const float*)d_in[0];
  const float* Wr = (const float*)d_in[1];
  const float* br = (const float*)d_in[2];
  const float* W1 = (const float*)d_in[3];
  const float* b1 = (const float*)d_in[4];
  const float* W2 = (const float*)d_in[5];
  const float* b2 = (const float*)d_in[6];
  float* out = (float*)d_out;

  char* ws = (char*)d_ws;
  unsigned short* W1p = (unsigned short*)(ws);                 // 4 MiB
  unsigned short* W2p = (unsigned short*)(ws + (4u << 20));    // 4 MiB
  size_t o = (8u << 20);
  float* gate   = (float*)(ws + o); o += N_TOK * 4;
  int*   idxArr = (int*)  (ws + o); o += N_TOK * 4;
  int*   perm   = (int*)  (ws + o); o += N_TOK * 4;
  int*   cnt    = (int*)  (ws + o); o += 64;
  int*   cursor = (int*)  (ws + o); o += 64;
  int*   off    = (int*)  (ws + o); o += 64;
  int*   tileOff= (int*)  (ws + o); o += 64;

  moe_init<<<1, 32, 0, stream>>>(cnt, cursor);
  moe_pack<<<16384, 256, 0, stream>>>(W1, W2, W1p, W2p);
  moe_router<<<N_TOK / 8, 256, 0, stream>>>(x, Wr, br, gate, idxArr, cnt);
  moe_scan<<<1, 32, 0, stream>>>(cnt, off, tileOff);
  moe_scatter<<<N_TOK / 256, 256, 0, stream>>>(idxArr, off, cursor, perm);
  // worst-case tile count: N/64 full tiles + up to E partial tiles
  moe_expert_gemm<<<N_TOK / 64 + NEXP, 256, 0, stream>>>(
      x, b1, b2, W1p, W2p, gate, perm, cnt, off, tileOff, out);
}
